// RouterLite_50525995270578
// MI455X (gfx1250) — matrix operand order repaired and checked by reading, not compiled, not run on hardware
//
#include <hip/hip_runtime.h>
#include <hip/hip_bf16.h>
#include <math.h>

// ---------------- CDNA5 WMMA plumbing (wave32) ----------------
typedef __attribute__((ext_vector_type(16))) _Float16 v16h;
typedef __attribute__((ext_vector_type(8)))  float    v8f;

static constexpr int Hdim = 2048;   // hidden
static constexpr int D2   = 256;    // 2*D
static constexpr int Dd   = 128;    // align dim

__device__ __forceinline__ v8f wmma_f16(v16h a, v16h b, v8f c) {
  // D = A(16x32,f16) * B(32x16,f16) + C(16x16,f32)
  return __builtin_amdgcn_wmma_f32_16x16x32_f16(false, a, false, b, (short)0, c,
                                                false, false);
}

// Scheduling fence: keep the fragment-load batch before the WMMA chain so all
// B fragments get distinct registers and the waits collapse to one.
__device__ __forceinline__ void sched_fence() {
#if __has_builtin(__builtin_amdgcn_sched_barrier)
  __builtin_amdgcn_sched_barrier(0);
#else
  asm volatile("" ::: "memory");
#endif
}

// A fragment 16x32 f16, row-major A[m][k] (ld halves, multiple of 8):
// lane<16 -> M=lane, K chunks {8h..8h+7} and {16+8h..23+8h} -> two 16B LDS loads.
__device__ __forceinline__ v16h load_frag_a_vec(const _Float16* base, int ld, int lane) {
  const int half = lane >> 4, m = lane & 15;
  const _Float16* p = base + m * ld + 8 * half;
  union { v16h v; uint4 q[2]; } u;
  u.q[0] = *(const uint4*)p;
  u.q[1] = *(const uint4*)(p + 16);
  return u.v;
}
// B fragment 32x16 f16 from COLUMN-major storage Bt[n][k] (ld halves, mult of 8):
// lane n holds K chunks {8h..8h+7} and {16+8h..23+8h} of column n -> two 16B LDS loads.
__device__ __forceinline__ v16h load_frag_b_cm(const _Float16* base, int ld, int lane) {
  const int half = lane >> 4, n = lane & 15;
  const _Float16* p = base + n * ld + 8 * half;
  union { v16h v; uint4 q[2]; } u;
  u.q[0] = *(const uint4*)p;
  u.q[1] = *(const uint4*)(p + 16);
  return u.v;
}

__device__ __forceinline__ float gelu_exact(float x) {
  return 0.5f * x * (1.0f + erff(x * 0.70710678118654752f));
}

union PackH4 { _Float16 h[4]; uint2 u2; };

// ---------------- K0a: plain f32 -> f16 downconvert (layout preserved) ----------
__global__ void k_f32_to_f16(const float* __restrict__ src,
                             _Float16* __restrict__ dst, int n) {
  for (int i = blockIdx.x * blockDim.x + threadIdx.x; i < n;
       i += gridDim.x * blockDim.x)
    dst[i] = (_Float16)src[i];
}

// ---------------- K0b: f16 transpose [R][C] -> [C][R] (for zrT) ----------------
__global__ void k_transpose_h(const _Float16* __restrict__ src,
                              _Float16* __restrict__ dst, int R, int C) {
  int total = R * C;
  for (int idx = blockIdx.x * blockDim.x + threadIdx.x; idx < total;
       idx += gridDim.x * blockDim.x) {
    int c = idx / R, r = idx - c * R;
    dst[idx] = src[(size_t)r * C + c];
  }
}

// ---------------- K1: fused LN -> W1^T -> GELU -> W2^T -> l2norm ----------------
// 64 rows per block, 256 threads (8 waves). Stage1 out 64x256, stage2 out 64x128.
union R2U {
  _Float16 h1[64][D2];   // 32 KB (GELU output, f16, ld=256)
  float    z[64][Dd];    // 32 KB (stage-2 fp32 result)
};

__global__ __launch_bounds__(256) void k_mlp_head(
    const float* __restrict__ X,  // [N][Hdim]
    const float* __restrict__ lng, const float* __restrict__ lnb,
    const _Float16* __restrict__ W1h,  // [256][Hdim]  f16, N-major (as reference)
    const float* __restrict__ b1,
    const _Float16* __restrict__ W2h,  // [128][256]   f16, N-major
    const float* __restrict__ b2,
    float* __restrict__ Zf,      // [N][128]
    _Float16* __restrict__ Zh) { // [N][128]
  __shared__ float s_mean[64], s_rstd[64];
  __shared__ __align__(16) _Float16 sA[64][40];    // LN'ed x tile 64x32 (ld=40) 5 KB
  __shared__ __align__(16) _Float16 sB1t[D2][32];  // W1 tile, column-major, 16 KB
  __shared__ __align__(16) R2U      sR2;           // 32 KB
  __shared__ __align__(16) _Float16 sB2t[Dd][32];  // W2 tile, column-major, 8 KB

  const int tid = threadIdx.x, wave = tid >> 5, lane = tid & 31;
  const int half = lane >> 4, nn = lane & 15;
  const int row0 = blockIdx.x * 64;

  // ---- phase A: per-row LN statistics (one row per wave, coalesced) ----
  for (int r = wave; r < 64; r += 8) {
    const float* xr = X + (size_t)(row0 + r) * Hdim;
    float s = 0.f, ss = 0.f;
    for (int c = lane; c < Hdim; c += 32) { float v = xr[c]; s += v; ss += v * v; }
#pragma unroll
    for (int m = 16; m > 0; m >>= 1) { s += __shfl_xor(s, m, 32); ss += __shfl_xor(ss, m, 32); }
    if (lane == 0) {
      float mean = s * (1.0f / Hdim);
      float var  = ss * (1.0f / Hdim) - mean * mean;
      s_mean[r] = mean;
      s_rstd[r] = rsqrtf(var + 1e-5f);
    }
  }
  __syncthreads();

  // ---- stage 1: (LN(x)) @ W1^T, 64x256, K=2048 ----
  const v8f vzero = {0.f, 0.f, 0.f, 0.f, 0.f, 0.f, 0.f, 0.f};
  v8f acc1[8];
#pragma unroll
  for (int t = 0; t < 8; ++t) acc1[t] = vzero;
  const int mtile = wave >> 1, nbase = (wave & 1) * 8;

  for (int k0 = 0; k0 < Hdim; k0 += 32) {
    __syncthreads();
    // A tile: LN applied on the fly; b128 loads, 4 packed halves per b64 store
    for (int e = tid; e < 64 * 8; e += 256) {
      int r = e >> 3, c = (e & 7) * 4;
      const float4 x4 = *(const float4*)(X + (size_t)(row0 + r) * Hdim + k0 + c);
      const float mu = s_mean[r], rs = s_rstd[r];
      PackH4 pk;
      pk.h[0] = (_Float16)((x4.x - mu) * rs * lng[k0 + c + 0] + lnb[k0 + c + 0]);
      pk.h[1] = (_Float16)((x4.y - mu) * rs * lng[k0 + c + 1] + lnb[k0 + c + 1]);
      pk.h[2] = (_Float16)((x4.z - mu) * rs * lng[k0 + c + 2] + lnb[k0 + c + 2]);
      pk.h[3] = (_Float16)((x4.w - mu) * rs * lng[k0 + c + 3] + lnb[k0 + c + 3]);
      *(uint2*)&sA[r][c] = pk.u2;
    }
    // B1 tile: W1 is N-major, so the column-major tile is a straight row slice.
    for (int e = tid; e < D2 * 4; e += 256) {  // 1024 x uint4 (8 halves)
      int n = e >> 2, kq = (e & 3) * 8;
      *(uint4*)&sB1t[n][kq] = *(const uint4*)(W1h + (size_t)n * Hdim + k0 + kq);
      if ((e & 3) == 0 && k0 + 32 < Hdim)
        __builtin_prefetch(W1h + (size_t)n * Hdim + k0 + 32, 0, 1);
    }
    __syncthreads();
    // batch: 1 A frag + 8 B frags, scheduling fence, then 8 back-to-back WMMAs
    v16h af = load_frag_a_vec(&sA[mtile * 16][0], 40, lane);
    v16h bf[8];
#pragma unroll
    for (int t = 0; t < 8; ++t)
      bf[t] = load_frag_b_cm(&sB1t[(nbase + t) * 16][0], 32, lane);
    sched_fence();
#pragma unroll
    for (int t = 0; t < 8; ++t) acc1[t] = wmma_f16(af, bf[t], acc1[t]);
  }

  // ---- bias + exact GELU, write h1 (f16) ----
  __syncthreads();
#pragma unroll
  for (int t = 0; t < 8; ++t) {
    int col = (nbase + t) * 16 + nn;
    float bb = b1[col];
#pragma unroll
    for (int i = 0; i < 8; ++i) {
      float hv = gelu_exact(acc1[t][i] + bb);
      sR2.h1[mtile * 16 + i + 8 * half][col] = (_Float16)hv;
    }
  }
  __syncthreads();

  // ---- stage 2: h1 @ W2^T, 64x128, K=256 ----
  v8f acc2[4];
#pragma unroll
  for (int t = 0; t < 4; ++t) acc2[t] = vzero;
  const int nbase2 = (wave & 1) * 4;

  for (int k0 = 0; k0 < D2; k0 += 32) {
    __syncthreads();
    for (int e = tid; e < Dd * 4; e += 256) {  // 512 x uint4
      int n = e >> 2, kq = (e & 3) * 8;
      *(uint4*)&sB2t[n][kq] = *(const uint4*)(W2h + (size_t)n * D2 + k0 + kq);
    }
    __syncthreads();
    v16h af = load_frag_a_vec(&sR2.h1[mtile * 16][k0], D2, lane);
    v16h bf[4];
#pragma unroll
    for (int t = 0; t < 4; ++t)
      bf[t] = load_frag_b_cm(&sB2t[(nbase2 + t) * 16][0], 32, lane);
    sched_fence();
#pragma unroll
    for (int t = 0; t < 4; ++t) acc2[t] = wmma_f16(af, bf[t], acc2[t]);
  }

  // ---- write z tile (fp32) then row-wise l2 normalize ----
  __syncthreads();
#pragma unroll
  for (int t = 0; t < 4; ++t) {
    int col = (nbase2 + t) * 16 + nn;
    float bb = b2[col];
#pragma unroll
    for (int i = 0; i < 8; ++i)
      sR2.z[mtile * 16 + i + 8 * half][col] = acc2[t][i] + bb;
  }
  __syncthreads();
  for (int r = tid; r < 64; r += 256) {
    float ss = 0.f;
    for (int c = 0; c < Dd; ++c) { float v = sR2.z[r][c]; ss += v * v; }
    float inv = 1.0f / fmaxf(sqrtf(ss), 1e-12f);
    size_t base = (size_t)(row0 + r) * Dd;
    for (int c = 0; c < Dd; ++c) {
      float v = sR2.z[r][c] * inv;
      Zf[base + c] = v;
      Zh[base + c] = (_Float16)v;
    }
  }
}

// ---------------- K2: flash-style scores+softmax+aggregation+entropy ----------------
// 128 query rows/block, 8 waves, each wave owns 16 rows. BN=32 key rows per step.
__global__ __launch_bounds__(256) void k_attn(
    const _Float16* __restrict__ Zq,   // [B][128] f16
    const _Float16* __restrict__ Zr,   // [L][128] f16 (key-major)
    const _Float16* __restrict__ ZrT,  // [128][L] f16 (dim-major)
    float* __restrict__ Rbar,          // [B][128]
    float* __restrict__ cosArr,        // [B]
    float* __restrict__ uArr,          // [B]
    int Ltot) {
  __shared__ __align__(16) _Float16 sQ[128][Dd];    // 32 KB
  __shared__ __align__(16) _Float16 sR[32][Dd];     // 8 KB  (key-major tile)
  __shared__ __align__(16) _Float16 sRt[Dd][32];    // 8 KB  (dim-major tile)
  __shared__ __align__(16) _Float16 sP[8][16][32];  // 8 KB  (per-wave P tile)

  const int tid = threadIdx.x, wave = tid >> 5, lane = tid & 31;
  const int half = lane >> 4, nn = lane & 15;
  const int row0 = blockIdx.x * 128;
  const float scale = 0.08838834764831845f;  // 1/sqrt(128)

  {  // vectorized load of the query tile
    const uint4* src = (const uint4*)(Zq + (size_t)row0 * Dd);
    uint4* dst = (uint4*)&sQ[0][0];
    for (int e = tid; e < 128 * Dd / 8; e += 256) dst[e] = src[e];
  }
  __syncthreads();

  // S-GEMM A fragments are loop-invariant: hoist them out of the L loop.
  v16h afQ[4];
#pragma unroll
  for (int kk = 0; kk < 4; ++kk)
    afQ[kk] = load_frag_a_vec(&sQ[wave * 16][kk * 32], Dd, lane);

  const v8f vzero = {0.f, 0.f, 0.f, 0.f, 0.f, 0.f, 0.f, 0.f};
  float rowmax[8], Zrow[8], Ssum[8];
  v8f O[8];
#pragma unroll
  for (int i = 0; i < 8; ++i) { rowmax[i] = -1e30f; Zrow[i] = 0.f; Ssum[i] = 0.f; O[i] = vzero; }

  for (int l0 = 0; l0 < Ltot; l0 += 32) {
    __syncthreads();
    {  // key-major tile: contiguous copy
      const uint4* src = (const uint4*)(Zr + (size_t)l0 * Dd);
      uint4* dst = (uint4*)&sR[0][0];
      for (int e = tid; e < 32 * Dd / 8; e += 256) dst[e] = src[e];
    }
    // dim-major tile: row slices of the globally pre-transposed ZrT
    for (int e = tid; e < Dd * 4; e += 256) {  // 512 x uint4
      int d = e >> 2, kq = (e & 3) * 8;
      *(uint4*)&sRt[d][kq] = *(const uint4*)(ZrT + (size_t)d * Ltot + l0 + kq);
    }
    __syncthreads();

    // S = zq(16 rows) @ zr^T (32 cols), K=128. sR[n][k] is column-major B storage.
    v8f S[2];
#pragma unroll
    for (int t = 0; t < 2; ++t) S[t] = vzero;
    {
      v16h bfS[8];
#pragma unroll
      for (int kk = 0; kk < 4; ++kk)
#pragma unroll
        for (int t = 0; t < 2; ++t)
          bfS[kk * 2 + t] = load_frag_b_cm(&sR[t * 16][kk * 32], Dd, lane);
      sched_fence();
#pragma unroll
      for (int kk = 0; kk < 4; ++kk)
#pragma unroll
        for (int t = 0; t < 2; ++t)
          S[t] = wmma_f16(afQ[kk], bfS[kk * 2 + t], S[t]);
    }

    // online softmax per row (row = i + 8*half inside this wave's 16 rows)
#pragma unroll
    for (int i = 0; i < 8; ++i) {
      float mx = -1e30f;
#pragma unroll
      for (int t = 0; t < 2; ++t) {
        float s = S[t][i] * scale;
        S[t][i] = s;
        mx = fmaxf(mx, s);
      }
#pragma unroll
      for (int m = 1; m < 16; m <<= 1) mx = fmaxf(mx, __shfl_xor(mx, m, 32));
      float mnew = fmaxf(rowmax[i], mx);
      float alpha = __expf(rowmax[i] - mnew);
      rowmax[i] = mnew;
      Zrow[i] *= alpha;
      Ssum[i] *= alpha;
#pragma unroll
      for (int t = 0; t < 8; ++t) O[t][i] = O[t][i] * alpha;
      float zp = 0.f, sp = 0.f;
#pragma unroll
      for (int t = 0; t < 2; ++t) {
        float s = S[t][i];
        float p = __expf(s - mnew);
        zp += p;
        sp += p * s;
        sP[wave][i + 8 * half][t * 16 + nn] = (_Float16)p;
      }
#pragma unroll
      for (int m = 1; m < 16; m <<= 1) { zp += __shfl_xor(zp, m, 32); sp += __shfl_xor(sp, m, 32); }
      Zrow[i] += zp;
      Ssum[i] += sp;
    }

    // O += P(16x32) @ zr(32x128). sRt[n][k] is column-major B storage.
    {
      v16h afP = load_frag_a_vec(&sP[wave][0][0], 32, lane);
      v16h bfO[8];
#pragma unroll
      for (int t = 0; t < 8; ++t)
        bfO[t] = load_frag_b_cm(&sRt[t * 16][0], 32, lane);
      sched_fence();
#pragma unroll
      for (int t = 0; t < 8; ++t) O[t] = wmma_f16(afP, bfO[t], O[t]);
    }
  }

  // finalize: r_bar = l2n(O); cos = zq . r_bar; u_n = entropy/log(L)
  const float invLogL = 1.0f / __logf((float)Ltot);
#pragma unroll
  for (int i = 0; i < 8; ++i) {
    float ss = 0.f;
#pragma unroll
    for (int t = 0; t < 8; ++t) ss += O[t][i] * O[t][i];
#pragma unroll
    for (int m = 1; m < 16; m <<= 1) ss += __shfl_xor(ss, m, 32);
    float inv = 1.0f / fmaxf(sqrtf(ss), 1e-12f);
    int lrow = wave * 16 + i + 8 * half;
    int grow = row0 + lrow;
    float cs = 0.f;
#pragma unroll
    for (int t = 0; t < 8; ++t) {
      float rb = O[t][i] * inv;
      Rbar[(size_t)grow * Dd + t * 16 + nn] = rb;
      cs += rb * (float)sQ[lrow][t * 16 + nn];
    }
#pragma unroll
    for (int m = 1; m < 16; m <<= 1) cs += __shfl_xor(cs, m, 32);
    if (nn == 0) {
      cosArr[grow] = cs;
      float ent = __logf(Zrow[i]) + rowmax[i] - Ssum[i] / Zrow[i];
      uArr[grow] = ent * invLogL;
    }
  }
}

// ---------------- K3: decision MLP + scalar heads ----------------
__global__ __launch_bounds__(128) void k_decide(
    const float* __restrict__ Zq, const float* __restrict__ Rbar,
    const float* __restrict__ cosArr, const float* __restrict__ uArr,
    const float* __restrict__ lng, const float* __restrict__ lnb,
    const float* __restrict__ W1d, const float* __restrict__ b1d,
    const float* __restrict__ W2d, const float* __restrict__ b2d,
    const float* __restrict__ beta, const float* __restrict__ W_len,
    const float* __restrict__ b_len, float* __restrict__ out, int Bn) {
  __shared__ float xs[256];
  __shared__ float rs[4], rq[4];
  const int row = blockIdx.x, t = threadIdx.x;
  const int lane = t & 31, wid = t >> 5;

  float a0 = Zq[(size_t)row * Dd + t];
  float a1 = Rbar[(size_t)row * Dd + t];
  float s = a0 + a1, q = a0 * a0 + a1 * a1;
#pragma unroll
  for (int m = 1; m < 32; m <<= 1) { s += __shfl_xor(s, m, 32); q += __shfl_xor(q, m, 32); }
  if (lane == 0) { rs[wid] = s; rq[wid] = q; }
  __syncthreads();
  s = rs[0] + rs[1] + rs[2] + rs[3];
  q = rq[0] + rq[1] + rq[2] + rq[3];
  float mean = s * (1.0f / 256.0f);
  float var  = q * (1.0f / 256.0f) - mean * mean;
  float rstd = rsqrtf(var + 1e-5f);
  xs[t]       = (a0 - mean) * rstd * lng[t] + lnb[t];
  xs[t + 128] = (a1 - mean) * rstd * lng[t + 128] + lnb[t + 128];
  __syncthreads();

  float acc = b1d[t];
  const float* wrow = W1d + (size_t)t * 256;
  for (int k = 0; k < 256; ++k) acc += xs[k] * wrow[k];
  float lp = gelu_exact(acc) * W2d[t];
#pragma unroll
  for (int m = 1; m < 32; m <<= 1) lp += __shfl_xor(lp, m, 32);
  __syncthreads();
  if (lane == 0) rs[wid] = lp;
  __syncthreads();
  if (t == 0) {
    float logit = rs[0] + rs[1] + rs[2] + rs[3] + b2d[0];
    out[row] = logit;
    float cs = cosArr[row], u = uArr[row];
    float dl = cs * W_len[0] + u * W_len[1] + b_len[0];
    float lf = tanhf(dl);  // TEMP_TAU = 1
    float dt = fminf(fmaxf(beta[0] * u + beta[1] * lf, -0.5f), 0.5f);
    out[Bn + row] = dt;
  }
}

// ---------------- host launch ----------------
extern "C" void kernel_launch(void* const* d_in, const int* in_sizes, int n_in,
                              void* d_out, int out_size, void* d_ws, size_t ws_size,
                              hipStream_t stream) {
  (void)n_in; (void)out_size; (void)ws_size;
  const float* q      = (const float*)d_in[0];
  const float* Rnon   = (const float*)d_in[1];
  const float* ln_q_g = (const float*)d_in[2];
  const float* ln_q_b = (const float*)d_in[3];
  const float* W1q    = (const float*)d_in[4];
  const float* b1q    = (const float*)d_in[5];
  const float* W2q    = (const float*)d_in[6];
  const float* b2q    = (const float*)d_in[7];
  const float* ln_r_g = (const float*)d_in[8];
  const float* ln_r_b = (const float*)d_in[9];
  const float* W1r    = (const float*)d_in[10];
  const float* b1r    = (const float*)d_in[11];
  const float* W2r    = (const float*)d_in[12];
  const float* b2r    = (const float*)d_in[13];
  const float* ln_d_g = (const float*)d_in[14];
  const float* ln_d_b = (const float*)d_in[15];
  const float* W1d    = (const float*)d_in[16];
  const float* b1d    = (const float*)d_in[17];
  const float* W2d    = (const float*)d_in[18];
  const float* b2d    = (const float*)d_in[19];
  const float* beta   = (const float*)d_in[20];
  const float* W_len  = (const float*)d_in[21];
  const float* b_len  = (const float*)d_in[22];
  float* out = (float*)d_out;

  const int B = in_sizes[0] / Hdim;
  const int L = in_sizes[1] / Hdim;

  char* ws = (char*)d_ws;
  size_t off = 0;
  auto take = [&](size_t bytes) {
    size_t o = off;
    off += (bytes + 255) & ~(size_t)255;
    return o;
  };
  _Float16* w1qH = (_Float16*)(ws + take((size_t)D2 * Hdim * 2));  // [256][2048]
  _Float16* w1rH = (_Float16*)(ws + take((size_t)D2 * Hdim * 2));
  _Float16* w2qH = (_Float16*)(ws + take((size_t)Dd * D2 * 2));    // [128][256]
  _Float16* w2rH = (_Float16*)(ws + take((size_t)Dd * D2 * 2));
  _Float16* zq_h = (_Float16*)(ws + take((size_t)B * Dd * 2));
  _Float16* zr_h = (_Float16*)(ws + take((size_t)L * Dd * 2));
  _Float16* zrT  = (_Float16*)(ws + take((size_t)Dd * L * 2));     // [128][L]
  float* zq_f    = (float*)(ws + take((size_t)B * Dd * 4));
  float* zr_f    = (float*)(ws + take((size_t)L * Dd * 4));
  float* rbar    = (float*)(ws + take((size_t)B * Dd * 4));
  float* cosArr  = (float*)(ws + take((size_t)B * 4));
  float* uArr    = (float*)(ws + take((size_t)B * 4));

  // K0: weight downconvert to f16 (layouts preserved; already N-major = tile-friendly)
  {
    int tot1 = D2 * Hdim;
    k_f32_to_f16<<<(tot1 + 255) / 256, 256, 0, stream>>>(W1q, w1qH, tot1);
    k_f32_to_f16<<<(tot1 + 255) / 256, 256, 0, stream>>>(W1r, w1rH, tot1);
    int tot2 = Dd * D2;
    k_f32_to_f16<<<(tot2 + 255) / 256, 256, 0, stream>>>(W2q, w2qH, tot2);
    k_f32_to_f16<<<(tot2 + 255) / 256, 256, 0, stream>>>(W2r, w2rH, tot2);
  }

  // K1: z_q and Z_r via fused WMMA MLP heads
  k_mlp_head<<<B / 64, 256, 0, stream>>>(q, ln_q_g, ln_q_b, w1qH, b1q, w2qH, b2q,
                                         zq_f, zq_h);
  k_mlp_head<<<L / 64, 256, 0, stream>>>(Rnon, ln_r_g, ln_r_b, w1rH, b1r, w2rH, b2r,
                                         zr_f, zr_h);

  // K0b: global transpose of Z_r (dim-major copy for the P@Zr B-operand)
  k_transpose_h<<<((Dd * L) + 255) / 256, 256, 0, stream>>>(zr_h, zrT, L, Dd);

  // K2: streaming softmax attention + entropy + cosine
  k_attn<<<B / 128, 256, 0, stream>>>(zq_h, zr_h, zrT, rbar, cosArr, uArr, L);

  // K3: decision MLP + scalar heads -> [raw_logit | delta_theta]
  k_decide<<<B, 128, 0, stream>>>(zq_f, rbar, cosArr, uArr, ln_d_g, ln_d_b, W1d, b1d,
                                  W2d, b2d, beta, W_len, b_len, out, B);
}
